// Attention_36112085024816
// MI455X (gfx1250) — compile-verified, matrix operand order checked by reading, not run on hardware
//
#include <hip/hip_runtime.h>
#include <hip/hip_bf16.h>

typedef __bf16 bf16_t;
typedef __attribute__((ext_vector_type(16))) __bf16 v16bf;
typedef __attribute__((ext_vector_type(8)))  __bf16 v8bf;
typedef __attribute__((ext_vector_type(4)))  __bf16 v4bf;
typedef __attribute__((ext_vector_type(8)))  float  v8f;
typedef __attribute__((ext_vector_type(4)))  float  v4f;
typedef unsigned int u32;
typedef __attribute__((ext_vector_type(4))) u32 uint32x4;
typedef __attribute__((ext_vector_type(8))) int int32x8;
typedef __attribute__((ext_vector_type(4))) int int32x4;

constexpr int B_   = 2;
constexpr int S_   = 1024;
constexpr int HID_ = 4096;
constexpr int NQ_  = 32;
constexpr int NG_  = 4;
constexpr int NKV_ = 8;
constexpr int D_   = 128;
constexpr int OFF_ = 1024;
constexpr int T_   = OFF_ + S_;          // 2048
constexpr float QSCALE = 0.08838834764831845f; // 1/sqrt(128)

__device__ __forceinline__ bf16_t to_bf16(float f) { return (bf16_t)f; }

__device__ __forceinline__ v16bf cat16(v8bf a, v8bf b) {
  return __builtin_shufflevector(a, b, 0,1,2,3,4,5,6,7,8,9,10,11,12,13,14,15);
}

// Load one 16-bit A/B fragment (per-lane K-chunks {half*8..+7} and {16+half*8..+7})
__device__ __forceinline__ v16bf loadfrag(const bf16_t* p) {
  const v8bf* q = reinterpret_cast<const v8bf*>(p);
  return cat16(q[0], q[2]);
}

__device__ __forceinline__ v8f wmma_bf16(v16bf a, v16bf b, v8f c) {
  return __builtin_amdgcn_wmma_f32_16x16x32_bf16(false, a, false, b, (short)0, c,
                                                 false, false);
}

// ---------------------------------------------------------------- f32 -> bf16
__global__ void cvt_f32_bf16(const float* __restrict__ src,
                             bf16_t* __restrict__ dst, int n4) {
  int i = blockIdx.x * blockDim.x + threadIdx.x;
  if (i >= n4) return;
  v4f v = reinterpret_cast<const v4f*>(src)[i];
  v4bf o = { to_bf16(v[0]), to_bf16(v[1]), to_bf16(v[2]), to_bf16(v[3]) };
  reinterpret_cast<v4bf*>(dst)[i] = o;
}

// --------------------------------------------------------- WMMA GEMM (bf16)
// out(M,N) = A(M,K) * W(N,K)^T + bias(N)
// Register-blocked: wave = 32x64 tile (2 M-tiles x 4 N-tiles, 8 WMMA / k-step).
// Block = 8 waves arranged 2x4 -> block tile 64x256.
// kvmode==1: scatter into (B,T,NQ,D) k/v region at t=OFF+s with NG replication.
__global__ __launch_bounds__(256)
void gemm_wmma(const bf16_t* __restrict__ A, const bf16_t* __restrict__ W,
               const float* __restrict__ bias, float* __restrict__ out,
               int M, int N, int K, int kvmode) {
  const int tid  = threadIdx.x;
  const int lane = tid & 31;
  const int wv   = tid >> 5;
  const int half = lane >> 4;
  const int l16  = lane & 15;
  const int m0 = blockIdx.y * 64 + (wv >> 2) * 32;
  const int n0 = blockIdx.x * 256 + (wv & 3) * 64;

  const bf16_t* arow0 = A + (size_t)(m0 + l16) * K;
  const bf16_t* arow1 = arow0 + (size_t)16 * K;
  const bf16_t* wrow0 = W + (size_t)(n0 + l16) * K;
  const bf16_t* wrow1 = wrow0 + (size_t)16 * K;
  const bf16_t* wrow2 = wrow0 + (size_t)32 * K;
  const bf16_t* wrow3 = wrow0 + (size_t)48 * K;

  v8f acc[2][4];
  for (int mi = 0; mi < 2; ++mi)
    for (int j = 0; j < 4; ++j)
      acc[mi][j] = (v8f){0.f,0.f,0.f,0.f,0.f,0.f,0.f,0.f};

  for (int k0 = 0; k0 < K; k0 += 32) {
    const int ko = k0 + half * 8;
    v16bf a0 = loadfrag(arow0 + ko);
    v16bf a1 = loadfrag(arow1 + ko);
    v16bf b0 = loadfrag(wrow0 + ko);
    v16bf b1 = loadfrag(wrow1 + ko);
    v16bf b2 = loadfrag(wrow2 + ko);
    v16bf b3 = loadfrag(wrow3 + ko);
    __builtin_prefetch(arow0 + k0 + 128, 0, 1);
    __builtin_prefetch(wrow0 + k0 + 128, 0, 1);
    acc[0][0] = wmma_bf16(a0, b0, acc[0][0]);
    acc[1][0] = wmma_bf16(a1, b0, acc[1][0]);
    acc[0][1] = wmma_bf16(a0, b1, acc[0][1]);
    acc[1][1] = wmma_bf16(a1, b1, acc[1][1]);
    acc[0][2] = wmma_bf16(a0, b2, acc[0][2]);
    acc[1][2] = wmma_bf16(a1, b2, acc[1][2]);
    acc[0][3] = wmma_bf16(a0, b3, acc[0][3]);
    acc[1][3] = wmma_bf16(a1, b3, acc[1][3]);
  }

  for (int j = 0; j < 4; ++j) {
    const int c  = n0 + j * 16 + l16;
    const float bv = bias ? bias[c] : 0.0f;
    if (!kvmode) {
      for (int mi = 0; mi < 2; ++mi)
        for (int r = 0; r < 8; ++r) {
          int m = m0 + mi * 16 + r + 8 * half;   // C layout: M = r + 8*half
          out[(size_t)m * N + c] = acc[mi][j][r] + bv;
        }
    } else {
      const int h = c >> 7;                      // kv head
      const int d = c & (D_ - 1);
      for (int mi = 0; mi < 2; ++mi)
        for (int r = 0; r < 8; ++r) {
          int m = m0 + mi * 16 + r + 8 * half;
          int b = m >> 10;                       // m / S
          int s = m & (S_ - 1);
          float val = acc[mi][j][r] + bv;
          size_t base = (((size_t)b * T_ + OFF_ + s) * NQ_ + h * NG_) * D_ + d;
          for (int g = 0; g < NG_; ++g) out[base + (size_t)g * D_] = val;
        }
    }
  }
}

// -------------------------------------------------------------------- RoPE
__global__ void rope_kernel(float* __restrict__ p, const float* __restrict__ ct,
                            const float* __restrict__ st, int tStride, int tOff,
                            float scale) {
  int i  = blockIdx.x * blockDim.x + threadIdx.x;
  int d2 = i & 63;
  int n  = (i >> 6) & (NQ_ - 1);
  int s  = (i >> 11) & (S_ - 1);
  int b  = i >> 21;
  size_t a = (((size_t)b * tStride + tOff + s) * NQ_ + n) * D_ + 2 * d2;
  float c = ct[s * 64 + d2], sn = st[s * 64 + d2];
  float xr = p[a], xi = p[a + 1];
  p[a]     = (xr * c - xi * sn) * scale;
  p[a + 1] = (xr * sn + xi * c) * scale;
}

// ------------------------------------------------------------- cache concat
__global__ void cache_copy(const float4* __restrict__ kc,
                           const float4* __restrict__ vc,
                           float4* __restrict__ ko, float4* __restrict__ vo) {
  int i = blockIdx.x * blockDim.x + threadIdx.x;   // < B*OFF*NQ*D/4
  const int per = OFF_ * NQ_ * D_ / 4;
  int b = i / per, r = i - b * per;
  size_t dst = (size_t)b * (T_ * NQ_ * D_ / 4) + r;
  ko[dst] = kc[i];
  vo[dst] = vc[i];
}

// ----------------------------------------------------- flash attention (WMMA)
// One block = (b, head n, 128-query tile); 8 waves, wave = 16 query rows.
// K tile staged via TDM tensor_load_to_lds; V tile staged transposed (bf16).
__global__ __launch_bounds__(256)
void attn_wmma(const float* __restrict__ qf, const bf16_t* __restrict__ kb16,
               const bf16_t* __restrict__ vb16, bf16_t* __restrict__ attn) {
  __shared__ bf16_t Kt[64][128];     // [key][d]              16 KB (TDM dest)
  __shared__ bf16_t Vt[128][64];     // [d][key] (transposed) 16 KB
  __shared__ bf16_t Ps[8][16][64];   // per-wave P scratch    16 KB

  const int tid  = threadIdx.x;
  const int lane = tid & 31;
  const int wv   = tid >> 5;
  const int half = lane >> 4;
  const int l16  = lane & 15;
  const int qb0  = blockIdx.x * 128;
  const int n    = blockIdx.y;
  const int b    = blockIdx.z;

  const u32 ldsKt = (u32)(uintptr_t)&Kt[0][0];

  // Q fragments (RoPE'd, pre-scaled by 1/sqrt(D)); convert f32->bf16 once.
  v16bf qfrag[4];
  {
    const float* qrow =
        qf + (((size_t)b * S_ + qb0 + wv * 16 + l16) * NQ_ + n) * D_;
    for (int kc = 0; kc < 4; ++kc) {
      const float* p0 = qrow + kc * 32 + half * 8;
      const float* p1 = p0 + 16;
      v8bf lo, hi;
      for (int j = 0; j < 8; ++j) { lo[j] = to_bf16(p0[j]); hi[j] = to_bf16(p1[j]); }
      qfrag[kc] = cat16(lo, hi);
    }
  }

  v8f o[8];
  for (int dn = 0; dn < 8; ++dn) o[dn] = (v8f){0.f,0.f,0.f,0.f,0.f,0.f,0.f,0.f};
  float mrow[8], lrow[8];
  for (int r = 0; r < 8; ++r) { mrow[r] = -3.0e38f; lrow[r] = 0.f; }

  const int kend = OFF_ + qb0 + 128;               // causal bound (block-uniform)
  for (int kb = 0; kb < kend; kb += 64) {
    // ---- K tile: one TDM descriptor moves 64x128 bf16 (rows strided NQ*D)
    if (wv == 0) {
      unsigned long long ga = (unsigned long long)(uintptr_t)
          (kb16 + (((size_t)b * T_ + kb) * NQ_ + n) * D_);
      // D# group 0: count=1 | lds_addr | global_addr[56:0] | type=2
      uint32x4 g0 = { 1u, ldsKt, (u32)ga,
                      ((u32)(ga >> 32) & 0x01FFFFFFu) | (2u << 30) };
      // D# group 1: wg_mask=0, data_size=1(2B); tensor_dim0=128, tensor_dim1=64;
      //             tile_dim0=128, tile_dim1=64; tensor_dim0_stride=NQ*D=4096
      int32x8 g1 = { (int)0x00010000u,        // data_size=2B
                     (int)(128u << 16),       // tensor_dim0[15:0]
                     (int)(64u  << 16),       // tensor_dim1[15:0]
                     (int)(128u << 16),       // tile_dim0
                     (int)64,                 // tile_dim1
                     (int)4096,               // tensor_dim0_stride lo
                     0, 0 };
      int32x4 gz4 = { 0, 0, 0, 0 };           // 2-D tensor: groups 2/3 unused
      int32x8 gz8 = { 0, 0, 0, 0, 0, 0, 0, 0 };
      __builtin_amdgcn_tensor_load_to_lds(g0, g1, gz4, gz4, gz8, 0);
      __builtin_amdgcn_s_wait_tensorcnt(0);
    }
    // ---- V tile: bf16 load + transpose into Vt[d][key]
    for (int idx = tid; idx < 64 * 16; idx += 256) {
      int row = idx >> 4, c8 = (idx & 15) * 8;
      const v8bf vv = *reinterpret_cast<const v8bf*>(
          vb16 + (((size_t)b * T_ + kb + row) * NQ_ + n) * D_ + c8);
      for (int j = 0; j < 8; ++j) Vt[c8 + j][row] = vv[j];
    }
    __syncthreads();

    // ---- S = Q * K^T  (4 key tiles of 16 x 4 K-steps of 32)
    v8f sc[4];
    for (int nt = 0; nt < 4; ++nt) {
      v8f a = {0.f,0.f,0.f,0.f,0.f,0.f,0.f,0.f};
      for (int kc = 0; kc < 4; ++kc)
        a = wmma_bf16(qfrag[kc],
                      loadfrag(&Kt[nt * 16 + l16][kc * 32 + half * 8]), a);
      sc[nt] = a;
    }

    // ---- causal mask
    for (int nt = 0; nt < 4; ++nt)
      for (int r = 0; r < 8; ++r) {
        int t  = kb + nt * 16 + l16;
        int sq = qb0 + wv * 16 + r + 8 * half;
        if (t > OFF_ + sq) sc[nt][r] = -1.0e30f;
      }

    // ---- online softmax (row stats per-lane for rows r+8*half)
    float scale[8];
    for (int r = 0; r < 8; ++r) {
      float v = fmaxf(fmaxf(sc[0][r], sc[1][r]), fmaxf(sc[2][r], sc[3][r]));
      for (int msk = 1; msk < 16; msk <<= 1) v = fmaxf(v, __shfl_xor(v, msk));
      float mn = fmaxf(mrow[r], v);
      scale[r] = __expf(mrow[r] - mn);
      mrow[r]  = mn;
    }
    float rsum[8];
    for (int r = 0; r < 8; ++r) rsum[r] = 0.f;
    for (int nt = 0; nt < 4; ++nt)
      for (int r = 0; r < 8; ++r) {
        float pexp = __expf(sc[nt][r] - mrow[r]);
        sc[nt][r] = pexp;
        rsum[r] += pexp;
      }
    for (int r = 0; r < 8; ++r) {
      float v = rsum[r];
      for (int msk = 1; msk < 16; msk <<= 1) v += __shfl_xor(v, msk);
      lrow[r] = lrow[r] * scale[r] + v;
    }
    for (int dn = 0; dn < 8; ++dn)
      for (int r = 0; r < 8; ++r) o[dn][r] *= scale[r];

    // ---- D-layout -> A-layout transpose of P through LDS
    for (int nt = 0; nt < 4; ++nt)
      for (int r = 0; r < 8; ++r)
        Ps[wv][r + 8 * half][nt * 16 + l16] = to_bf16(sc[nt][r]);
    __syncthreads();

    // ---- O += P * V
    for (int kc2 = 0; kc2 < 2; ++kc2) {
      v16bf ap = loadfrag(&Ps[wv][l16][kc2 * 32 + half * 8]);
      for (int dn = 0; dn < 8; ++dn)
        o[dn] = wmma_bf16(ap,
                          loadfrag(&Vt[dn * 16 + l16][kc2 * 32 + half * 8]),
                          o[dn]);
    }
    __syncthreads();
  }

  // ---- normalize and emit bf16 for the O-projection GEMM
  for (int dn = 0; dn < 8; ++dn)
    for (int r = 0; r < 8; ++r) {
      int sq = qb0 + wv * 16 + r + 8 * half;
      float val = o[dn][r] / lrow[r];
      attn[(((size_t)b * S_ + sq) * NQ_ + n) * D_ + dn * 16 + l16] = to_bf16(val);
    }
}

// --------------------------------------------------------------------- host
extern "C" void kernel_launch(void* const* d_in, const int* in_sizes, int n_in,
                              void* d_out, int out_size, void* d_ws,
                              size_t ws_size, hipStream_t stream) {
  (void)in_sizes; (void)n_in; (void)out_size; (void)ws_size;
  const float* x      = (const float*)d_in[0];
  const float* fcos   = (const float*)d_in[2];
  const float* fsin   = (const float*)d_in[3];
  const float* kcache = (const float*)d_in[4];
  const float* vcache = (const float*)d_in[5];
  const float* q_w    = (const float*)d_in[6];
  const float* q_b    = (const float*)d_in[7];
  const float* k_w    = (const float*)d_in[8];
  const float* k_b    = (const float*)d_in[9];
  const float* v_w    = (const float*)d_in[10];
  const float* v_b    = (const float*)d_in[11];
  const float* o_w    = (const float*)d_in[12];

  float* out  = (float*)d_out;                          // (B,S,NQ*D)
  float* kout = out  + (size_t)B_ * S_ * NQ_ * D_;      // (B,T,NQ,D)
  float* vout = kout + (size_t)B_ * T_ * NQ_ * D_;      // (B,T,NQ,D)

  char* ws = (char*)d_ws;
  const size_t MB = (size_t)1 << 20;
  bf16_t* xb   = (bf16_t*)(ws);              //  16 MiB  x bf16
  bf16_t* qwb  = (bf16_t*)(ws + 16  * MB);   //  32 MiB
  bf16_t* kwb  = (bf16_t*)(ws + 48  * MB);   //   8 MiB
  bf16_t* vwb  = (bf16_t*)(ws + 56  * MB);   //   8 MiB
  bf16_t* owb  = (bf16_t*)(ws + 64  * MB);   //  32 MiB
  float*  qfb  = (float*) (ws + 96  * MB);   //  32 MiB  Q f32 (rope'd, scaled)
  bf16_t* ab   = (bf16_t*)(ws + 128 * MB);   //  16 MiB  attn out bf16
  bf16_t* kb16 = (bf16_t*)(ws + 144 * MB);   //  32 MiB  K concat bf16
  bf16_t* vb16 = (bf16_t*)(ws + 176 * MB);   //  32 MiB  V concat bf16

  auto cvt = [&](const float* s, bf16_t* d, int nelem) {
    int n4 = nelem / 4;
    cvt_f32_bf16<<<dim3((n4 + 255) / 256), dim3(256), 0, stream>>>(s, d, n4);
  };
  cvt(x,   xb,  B_ * S_ * HID_);
  cvt(q_w, qwb, NQ_  * D_ * HID_);
  cvt(k_w, kwb, NKV_ * D_ * HID_);
  cvt(v_w, vwb, NKV_ * D_ * HID_);
  cvt(o_w, owb, HID_ * NQ_ * D_);

  const int M = B_ * S_;
  gemm_wmma<<<dim3(4096 / 256, M / 64), 256, 0, stream>>>(
      xb, qwb, q_b, qfb, M, 4096, HID_, 0);
  gemm_wmma<<<dim3(1024 / 256, M / 64), 256, 0, stream>>>(
      xb, kwb, k_b, kout, M, 1024, HID_, 1);
  gemm_wmma<<<dim3(1024 / 256, M / 64), 256, 0, stream>>>(
      xb, vwb, v_b, vout, M, 1024, HID_, 1);

  cache_copy<<<dim3(B_ * OFF_ * NQ_ * D_ / 4 / 256), 256, 0, stream>>>(
      (const float4*)kcache, (const float4*)vcache, (float4*)kout, (float4*)vout);

  const int npairs = B_ * S_ * NQ_ * (D_ / 2);
  rope_kernel<<<dim3(npairs / 256), 256, 0, stream>>>(qfb, fcos, fsin, S_, 0, QSCALE);
  rope_kernel<<<dim3(npairs / 256), 256, 0, stream>>>(kout, fcos, fsin, T_, OFF_, 1.0f);

  // one-time bf16 copies of concatenated K/V for the WMMA attention pipe
  cvt(kout, kb16, B_ * T_ * NQ_ * D_);
  cvt(vout, vb16, B_ * T_ * NQ_ * D_);

  attn_wmma<<<dim3(S_ / 128, NQ_, B_), 256, 0, stream>>>(qfb, kb16, vb16, ab);

  gemm_wmma<<<dim3(4096 / 256, M / 64), 256, 0, stream>>>(
      ab, owb, nullptr, out, M, HID_, 4096, 0);
}
